// MultiHeadAttention_13365938225590
// MI455X (gfx1250) — compile-verified, matrix operand order checked by reading, not run on hardware
//
#include <hip/hip_runtime.h>
#include <hip/hip_bf16.h>
#include <stdint.h>

// ---- problem constants ----
#define BB 2
#define LL 2048
#define SS 2048
#define DD 1024
#define HH 16
#define HD 64
#define SCALE 0.125f       // HD^-0.5
#define NEGV -1.0e9f

typedef __attribute__((ext_vector_type(16))) __bf16 bf16x16;
typedef __attribute__((ext_vector_type(4)))  __bf16 bf16x4;
typedef __attribute__((ext_vector_type(8)))  float  f32x8;

struct Frag { union { bf16x16 v; uint4 u[2]; }; };

__device__ __forceinline__ f32x8 zero8() {
    f32x8 z = {0.f,0.f,0.f,0.f,0.f,0.f,0.f,0.f};
    return z;
}

__device__ __forceinline__ f32x8 wmma_bf16(const Frag& a, const Frag& b, f32x8 c) {
    return __builtin_amdgcn_wmma_f32_16x16x32_bf16(
        /*neg_a=*/false, a.v, /*neg_b=*/false, b.v,
        /*c_mod=*/(short)0, c, /*reuse_a=*/false, /*reuse_b=*/false);
}

// LDS byte offset from a generic pointer into __shared__ (flat addr low 32 bits)
__device__ __forceinline__ unsigned as_lds(const void* p) {
    return (unsigned)(unsigned long long)p;
}

// async 16-byte global -> LDS copy (per enabled lane), tracked by ASYNCcnt
__device__ __forceinline__ void async_ld16(unsigned lds, const void* g) {
    asm volatile("global_load_async_to_lds_b128 %0, %1, off"
                 :: "v"(lds), "v"(g) : "memory");
}

__device__ __forceinline__ void wait_async0() {
    asm volatile("s_wait_asynccnt 0x0" ::: "memory");
}

// ---------------------------------------------------------------------------
// f32 -> bf16 converter (vectorized x4)
// ---------------------------------------------------------------------------
__global__ void cvt_f32_to_bf16(const float* __restrict__ x,
                                __bf16* __restrict__ y, int n4) {
    int i = blockIdx.x * blockDim.x + threadIdx.x;
    if (i >= n4) return;
    float4 f = *(const float4*)(x + (size_t)i * 4);
    bf16x4 o = { (__bf16)f.x, (__bf16)f.y, (__bf16)f.z, (__bf16)f.w };
    *(bf16x4*)(y + (size_t)i * 4) = o;
}

// ---------------------------------------------------------------------------
// WMMA GEMM:  Y[m,n] = sum_k A[m,k] * W[n,k]    (A: MxK bf16, W: NxK bf16)
// Block: 256 thr (8 waves), tile 128x128. Wave tile 64x32 = 4x2 wmma tiles.
// mode 0: f32 plain [M,N]
// mode 1: bf16 head-split  Out[b,h,l,hd]
// mode 2: bf16 head-split transposed Out[b,h,hd,s] (for V)
// ---------------------------------------------------------------------------
__global__ __launch_bounds__(256)
void gemm_wmma_nt(const __bf16* __restrict__ A, const __bf16* __restrict__ W,
                  void* __restrict__ Out, int M, int N, int K, int mode) {
    const int lane = threadIdx.x & 31;
    const int wave = threadIdx.x >> 5;
    const int ln   = lane & 15;
    const int half = lane >> 4;
    const int wm   = wave & 1;
    const int wn   = wave >> 1;
    const int mb   = blockIdx.x * 128 + wm * 64;
    const int nb   = blockIdx.y * 128 + wn * 32;

    f32x8 acc[4][2];
    for (int i = 0; i < 4; ++i)
        for (int j = 0; j < 2; ++j) acc[i][j] = zero8();

    for (int kb = 0; kb < K; kb += 32) {
        const int kl = kb + half * 8;
        Frag a[4];
#pragma unroll
        for (int i = 0; i < 4; ++i) {
            const __bf16* p = A + (size_t)(mb + i * 16 + ln) * K + kl;
            a[i].u[0] = *(const uint4*)p;
            a[i].u[1] = *(const uint4*)(p + 16);
            __builtin_prefetch(p + 32, 0, 1);
        }
        Frag b[2];
#pragma unroll
        for (int j = 0; j < 2; ++j) {
            const __bf16* p = W + (size_t)(nb + j * 16 + ln) * K + kl;
            b[j].u[0] = *(const uint4*)p;
            b[j].u[1] = *(const uint4*)(p + 16);
            __builtin_prefetch(p + 32, 0, 1);
        }
#pragma unroll
        for (int i = 0; i < 4; ++i)
#pragma unroll
            for (int j = 0; j < 2; ++j)
                acc[i][j] = wmma_bf16(a[i], b[j], acc[i][j]);
    }

#pragma unroll
    for (int i = 0; i < 4; ++i) {
#pragma unroll
        for (int j = 0; j < 2; ++j) {
            const int n = nb + j * 16 + ln;
#pragma unroll
            for (int r = 0; r < 8; ++r) {
                const int m = mb + i * 16 + r + half * 8;
                const float val = acc[i][j][r];
                if (mode == 0) {
                    ((float*)Out)[(size_t)m * N + n] = val;
                } else if (mode == 1) {
                    const int bidx = m >> 11, l = m & (LL - 1);
                    const int h = n >> 6, hd = n & (HD - 1);
                    ((__bf16*)Out)[((((size_t)bidx * HH + h) * LL) + l) * HD + hd] =
                        (__bf16)val;
                } else {
                    const int bidx = m >> 11, s = m & (SS - 1);
                    const int h = n >> 6, hd = n & (HD - 1);
                    ((__bf16*)Out)[((((size_t)bidx * HH + h) * HD) + hd) * SS + s] =
                        (__bf16)val;
                }
            }
        }
    }
}

// ---------------------------------------------------------------------------
// Flash attention per (b,h). Block = 128 thr (4 waves); each wave owns a
// 16-row L tile. K/V chunks (64x64 bf16) are cooperatively staged into LDS
// with async global->LDS copies, double-buffered to overlap with WMMA.
// ---------------------------------------------------------------------------
__device__ __forceinline__ void stage_kv(const __bf16* Kp, const __bf16* Vp, int sb,
                                         __bf16* kdst, __bf16* vdst,
                                         int wave, int lane) {
    // K chunk: 64 rows x 128B contiguous (8KB); wave covers its 2KB quarter
    const char* kg = (const char*)(Kp + (size_t)sb * HD);
    char* kl = (char*)kdst;
    const unsigned kb = wave * 2048 + lane * 16;
#pragma unroll
    for (int i = 0; i < 4; ++i)
        async_ld16(as_lds(kl + kb + i * 512), kg + kb + i * 512);
    // V chunk: 64 rows (hd) x 128B, global row stride SS*2 bytes
    const int rsub = wave * 4 + (lane >> 3);
    const int cb   = (lane & 7) * 16;
#pragma unroll
    for (int i = 0; i < 4; ++i) {
        const int r = i * 16 + rsub;
        async_ld16(as_lds((char*)vdst + r * 128 + cb),
                   (const char*)Vp + (size_t)r * (SS * 2) + (size_t)sb * 2 + cb);
    }
}

__global__ __launch_bounds__(128)
void attn_flash_wmma(const __bf16* __restrict__ Qh, const __bf16* __restrict__ Kh,
                     const __bf16* __restrict__ Vt, const uint8_t* __restrict__ mask,
                     __bf16* __restrict__ ctx) {
    __shared__ __align__(16) __bf16 Pbuf[4][16 * 64];   // per-wave private P tile
    __shared__ __align__(16) __bf16 Ksh[2][64 * 64];    // staged K chunk [s][hd]
    __shared__ __align__(16) __bf16 Vsh[2][64 * 64];    // staged V chunk [hd][s]

    const int lane = threadIdx.x & 31;
    const int wave = threadIdx.x >> 5;
    const int ln   = lane & 15;
    const int half = lane >> 4;
    const int b    = blockIdx.z;
    const int h    = blockIdx.y;
    const int l0   = blockIdx.x * 64 + wave * 16;

    const size_t bh = (size_t)b * HH + h;
    const __bf16*  Qp = Qh + (bh * LL + l0) * HD;
    const __bf16*  Kp = Kh + bh * SS * HD;
    const __bf16*  Vp = Vt + bh * HD * SS;
    const uint8_t* mp = mask + ((size_t)b * LL + l0) * SS;
    __bf16* Pw = &Pbuf[wave][0];

    // Q fragments (K dim = HD = 64 -> 2 k-steps of 32), loaded once
    Frag qf[2];
#pragma unroll
    for (int j = 0; j < 2; ++j) {
        const __bf16* p = Qp + (size_t)ln * HD + j * 32 + half * 8;
        qf[j].u[0] = *(const uint4*)p;
        qf[j].u[1] = *(const uint4*)(p + 16);
    }

    f32x8 acc[4];
#pragma unroll
    for (int t = 0; t < 4; ++t) acc[t] = zero8();
    float rmax[8], rsum[8];
#pragma unroll
    for (int i = 0; i < 8; ++i) { rmax[i] = -1.0e30f; rsum[i] = 0.f; }

    // prologue: fill buffer 0
    stage_kv(Kp, Vp, 0, Ksh[0], Vsh[0], wave, lane);

    int buf = 0;
    for (int sb = 0; sb < SS; sb += 64, buf ^= 1) {
        wait_async0();          // this wave's fill of `buf` complete
        __syncthreads();        // all waves' quarters visible
        if (sb + 64 < SS)
            stage_kv(Kp, Vp, sb + 64, Ksh[buf ^ 1], Vsh[buf ^ 1], wave, lane);

        const __bf16* Kb = &Ksh[buf][0];
        const __bf16* Vb = &Vsh[buf][0];

        // ---- scores tile: 16(L) x 64(S) ----
        f32x8 sc[4];
#pragma unroll
        for (int t = 0; t < 4; ++t) sc[t] = zero8();
#pragma unroll
        for (int j = 0; j < 2; ++j) {
#pragma unroll
            for (int t = 0; t < 4; ++t) {
                Frag kf;
                const __bf16* p = Kb + (size_t)(t * 16 + ln) * HD + j * 32 + half * 8;
                kf.u[0] = *(const uint4*)p;
                kf.u[1] = *(const uint4*)(p + 16);
                sc[t] = wmma_bf16(qf[j], kf, sc[t]);
            }
        }

        // ---- online softmax over this 64-wide chunk ----
#pragma unroll
        for (int i = 0; i < 8; ++i) {
            const int r = i + half * 8;
            float x[4];
            float cm = -1.0e30f;
#pragma unroll
            for (int t = 0; t < 4; ++t) {
                float xv = sc[t][i] * SCALE;
                const uint8_t mv = mp[(size_t)r * SS + sb + t * 16 + ln];
                if (mv) xv = NEGV;
                x[t] = xv;
                cm = fmaxf(cm, xv);
            }
#pragma unroll
            for (int d = 1; d < 16; d <<= 1)
                cm = fmaxf(cm, __shfl_xor(cm, d, 32));
            const float nm   = fmaxf(rmax[i], cm);
            const float resc = __expf(rmax[i] - nm);
            rmax[i] = nm;
            float ps = 0.f;
#pragma unroll
            for (int t = 0; t < 4; ++t) {
                const float pv = __expf(x[t] - nm);
                ps += pv;
                Pw[(size_t)r * 64 + t * 16 + ln] = (__bf16)pv;
            }
#pragma unroll
            for (int d = 1; d < 16; d <<= 1)
                ps += __shfl_xor(ps, d, 32);
            rsum[i] = rsum[i] * resc + ps;
#pragma unroll
            for (int t = 0; t < 4; ++t) acc[t][i] *= resc;
        }
        __builtin_amdgcn_wave_barrier();   // keep P stores before P loads

        // ---- ctx += P @ V (B operand from staged Vsh rows) ----
#pragma unroll
        for (int j = 0; j < 2; ++j) {
            Frag pf;
            const __bf16* pp = Pw + (size_t)ln * 64 + j * 32 + half * 8;
            pf.u[0] = *(const uint4*)pp;
            pf.u[1] = *(const uint4*)(pp + 16);
#pragma unroll
            for (int t = 0; t < 4; ++t) {
                Frag vf;
                const __bf16* p = Vb + (size_t)(t * 16 + ln) * 64 + j * 32 + half * 8;
                vf.u[0] = *(const uint4*)p;
                vf.u[1] = *(const uint4*)(p + 16);
                acc[t] = wmma_bf16(pf, vf, acc[t]);
            }
        }
        __syncthreads();   // all waves done reading `buf` before its refill
    }

    // ---- epilogue: normalize and store merged-head ctx (bf16) ----
#pragma unroll
    for (int i = 0; i < 8; ++i) {
        const int r = i + half * 8;
        const float inv = 1.0f / rsum[i];
#pragma unroll
        for (int t = 0; t < 4; ++t) {
            ctx[((size_t)b * LL + l0 + r) * DD + h * HD + t * 16 + ln] =
                (__bf16)(acc[t][i] * inv);
        }
    }
}

// ---------------------------------------------------------------------------
extern "C" void kernel_launch(void* const* d_in, const int* in_sizes, int n_in,
                              void* d_out, int out_size, void* d_ws, size_t ws_size,
                              hipStream_t stream) {
    (void)in_sizes; (void)n_in; (void)out_size; (void)ws_size;
    const float*   q    = (const float*)d_in[0];
    const float*   k    = (const float*)d_in[1];
    const float*   v    = (const float*)d_in[2];
    const uint8_t* mask = (const uint8_t*)d_in[3];
    const float*   Wq   = (const float*)d_in[4];
    const float*   Wk   = (const float*)d_in[5];
    const float*   Wv   = (const float*)d_in[6];
    const float*   Wo   = (const float*)d_in[7];
    float* out = (float*)d_out;

    const size_t NX = (size_t)BB * LL * DD;   // 4,194,304
    const size_t NW = (size_t)DD * DD;        // 1,048,576

    char* ws = (char*)d_ws;
    size_t off = 0;
    __bf16* qb   = (__bf16*)(ws + off); off += NX * 2;
    __bf16* kb   = (__bf16*)(ws + off); off += NX * 2;
    __bf16* vb   = (__bf16*)(ws + off); off += NX * 2;
    __bf16* wqb  = (__bf16*)(ws + off); off += NW * 2;
    __bf16* wkb  = (__bf16*)(ws + off); off += NW * 2;
    __bf16* wvb  = (__bf16*)(ws + off); off += NW * 2;
    __bf16* wob  = (__bf16*)(ws + off); off += NW * 2;
    __bf16* Qhb  = (__bf16*)(ws + off); off += NX * 2;
    __bf16* Khb  = (__bf16*)(ws + off); off += NX * 2;
    __bf16* Vtb  = (__bf16*)(ws + off); off += NX * 2;
    __bf16* ctxb = (__bf16*)(ws + off); off += NX * 2;

    const int CB = 256;
    cvt_f32_to_bf16<<<(int)(NX / 4 + CB - 1) / CB, CB, 0, stream>>>(q,  qb,  (int)(NX / 4));
    cvt_f32_to_bf16<<<(int)(NX / 4 + CB - 1) / CB, CB, 0, stream>>>(k,  kb,  (int)(NX / 4));
    cvt_f32_to_bf16<<<(int)(NX / 4 + CB - 1) / CB, CB, 0, stream>>>(v,  vb,  (int)(NX / 4));
    cvt_f32_to_bf16<<<(int)(NW / 4 + CB - 1) / CB, CB, 0, stream>>>(Wq, wqb, (int)(NW / 4));
    cvt_f32_to_bf16<<<(int)(NW / 4 + CB - 1) / CB, CB, 0, stream>>>(Wk, wkb, (int)(NW / 4));
    cvt_f32_to_bf16<<<(int)(NW / 4 + CB - 1) / CB, CB, 0, stream>>>(Wv, wvb, (int)(NW / 4));
    cvt_f32_to_bf16<<<(int)(NW / 4 + CB - 1) / CB, CB, 0, stream>>>(Wo, wob, (int)(NW / 4));

    const int M = BB * LL;          // 4096
    dim3 gg(M / 128, DD / 128);     // (32, 8)
    gemm_wmma_nt<<<gg, 256, 0, stream>>>(qb,   wqb, (void*)Qhb, M, DD, DD, 1);
    gemm_wmma_nt<<<gg, 256, 0, stream>>>(kb,   wkb, (void*)Khb, M, DD, DD, 1);
    gemm_wmma_nt<<<gg, 256, 0, stream>>>(vb,   wvb, (void*)Vtb, M, DD, DD, 2);

    dim3 ga(LL / 64, HH, BB);       // (32, 16, 2)
    attn_flash_wmma<<<ga, 128, 0, stream>>>(Qhb, Khb, Vtb, mask, ctxb);

    gemm_wmma_nt<<<gg, 256, 0, stream>>>(ctxb, wob, (void*)out, M, DD, DD, 0);
}
